// GCN_2903397892152
// MI455X (gfx1250) — matrix operand order repaired and checked by reading, not compiled, not run on hardware
//
#include <hip/hip_runtime.h>
#include <hip/hip_bf16.h>
#include <math.h>

#define N_NODES 50000
#define N_EDGES 800000
#define NCLS    10
#define EPS_F   1e-5f

typedef __attribute__((ext_vector_type(16))) __bf16 v16bf;
typedef __attribute__((ext_vector_type(8)))  float  v8f;

// ---------- helpers ----------
static __device__ __forceinline__ unsigned short f2bf(float x) {
  unsigned u = __float_as_uint(x);
  unsigned r = ((u >> 16) & 1u) + 0x7FFFu;   // round-to-nearest-even
  return (unsigned short)((u + r) >> 16);
}

// float atomic max/min via monotonic int encoding (mx init = -inf, mn init = +inf)
static __device__ __forceinline__ void atomicMaxF(float* a, float v) {
  if (v >= 0.0f) atomicMax((int*)a, __float_as_int(v));
  else           atomicMin((unsigned int*)a, __float_as_uint(v));
}
static __device__ __forceinline__ void atomicMinF(float* a, float v) {
  if (v >= 0.0f) atomicMin((int*)a, __float_as_int(v));
  else           atomicMax((unsigned int*)a, __float_as_uint(v));
}

// ---------- init ----------
__global__ void k_zero_f(float* p, long n) {
  long i = (long)blockIdx.x * blockDim.x + threadIdx.x;
  if (i < n) p[i] = 0.0f;
}
__global__ void k_fill_u(unsigned* p, long n, unsigned v) {
  long i = (long)blockIdx.x * blockDim.x + threadIdx.x;
  if (i < n) p[i] = v;
}

// ---------- degree + delta ----------
__global__ void k_deg(const int* __restrict__ dst, float* __restrict__ deg) {
  int e = blockIdx.x * blockDim.x + threadIdx.x;
  if (e < N_EDGES) atomicAdd(&deg[dst[e]], 1.0f);
}
__global__ void k_logdeg(const float* __restrict__ deg, float* __restrict__ logdeg,
                         float* __restrict__ dsum) {
  __shared__ float sred[256];
  int i = blockIdx.x * blockDim.x + threadIdx.x;
  float ld = 0.0f;
  if (i < N_NODES) { ld = log1pf(deg[i]); logdeg[i] = ld; }
  sred[threadIdx.x] = ld;
  __syncthreads();
  for (int o = blockDim.x >> 1; o > 0; o >>= 1) {
    if (threadIdx.x < o) sred[threadIdx.x] += sred[threadIdx.x + o];
    __syncthreads();
  }
  if (threadIdx.x == 0) atomicAdd(dsum, sred[0]);
}

// ---------- edge scatter: sum / sumsq / min / max (d is a power of two) ----------
__global__ void k_scatter(const float* __restrict__ h,
                          const int* __restrict__ src, const int* __restrict__ dst,
                          int sh, int d,
                          float* __restrict__ sum, float* __restrict__ sumsq,
                          float* __restrict__ mn, float* __restrict__ mx) {
  long idx = (long)blockIdx.x * blockDim.x + threadIdx.x;
  long total = (long)N_EDGES << sh;
  if (idx >= total) return;
  int e = (int)(idx >> sh);
  int f = (int)(idx & (d - 1));
  int s = src[e], t = dst[e];
  float v = h[(long)s * d + f];
  long o = (long)t * d + f;
  atomicAdd(&sum[o], v);
  atomicAdd(&sumsq[o], v * v);
  atomicMaxF(&mx[o], v);
  atomicMinF(&mn[o], v);
}

// ---------- build 13d PNA features (bf16, row-major [N, 13d]) ----------
__global__ void k_build_feat(const float* __restrict__ h,
                             const float* __restrict__ deg,
                             const float* __restrict__ logdeg,
                             const float* __restrict__ dsum,
                             int sh, int d, int ldf,
                             const float* __restrict__ sum, const float* __restrict__ sumsq,
                             const float* __restrict__ mn, const float* __restrict__ mx,
                             unsigned short* __restrict__ feat) {
  long idx = (long)blockIdx.x * blockDim.x + threadIdx.x;
  long total = (long)N_NODES << sh;
  if (idx >= total) return;
  int i = (int)(idx >> sh);
  int f = (int)(idx & (d - 1));
  float dg  = deg[i];
  float cnt = fmaxf(dg, 1.0f);
  long  o   = (long)i * d + f;
  float mean = sum[o] / cnt;
  float msq  = sumsq[o] / cnt;
  float sd   = sqrtf(fmaxf(msq - mean * mean, 0.0f) + EPS_F);
  bool  has  = dg > 0.0f;
  float mnv  = has ? mn[o] : 0.0f;
  float mxv  = has ? mx[o] : 0.0f;
  float ld    = logdeg[i];
  float delta = dsum[0] * (1.0f / (float)N_NODES);
  float amp   = ld / delta;
  float att   = (ld > 0.0f) ? (delta / fmaxf(ld, 1e-6f)) : 1.0f;
  unsigned short* row = feat + (long)i * ldf;
  row[f]           = f2bf(h[o]);
  row[d + f]       = f2bf(mean);
  row[2 * d + f]   = f2bf(mnv);
  row[3 * d + f]   = f2bf(mxv);
  row[4 * d + f]   = f2bf(sd);
  row[5 * d + f]   = f2bf(mean * amp);
  row[6 * d + f]   = f2bf(mnv * amp);
  row[7 * d + f]   = f2bf(mxv * amp);
  row[8 * d + f]   = f2bf(sd * amp);
  row[9 * d + f]   = f2bf(mean * att);
  row[10 * d + f]  = f2bf(mnv * att);
  row[11 * d + f]  = f2bf(mxv * att);
  row[12 * d + f]  = f2bf(sd * att);
}

// ---------- pack W [K,Nout] f32 -> B-fragment-ordered bf16 pairs ----------
// Fragment layout (wave32, 16-bit B, 32x16 per tile): lane l -> n = l&15,
// hi = l>>4; dword v (0..7) holds K = ktile*32 + (v&4?16:0) + (v&3)*2 + hi*8 (+1).
__global__ void k_packW(const float* __restrict__ W, int K, int Nout,
                        unsigned* __restrict__ Wp) {
  long idx = (long)blockIdx.x * blockDim.x + threadIdx.x;
  long total = (long)K * Nout / 2;
  if (idx >= total) return;
  int v = (int)(idx & 7);
  long t = idx >> 3;
  int lane = (int)(t & 31); t >>= 5;
  int nT = Nout >> 4;
  int nt = (int)(t % nT);
  int kt = (int)(t / nT);
  int n  = nt * 16 + (lane & 15);
  int hi = lane >> 4;
  int k  = kt * 32 + ((v & 4) ? 16 : 0) + (v & 3) * 2 + hi * 8;
  unsigned lo  = f2bf(W[(long)k * Nout + n]);
  unsigned hib = f2bf(W[(long)(k + 1) * Nout + n]);
  Wp[idx] = lo | (hib << 16);
}

// ---------- WMMA GEMM: out[N,Nout] = relu(feat[N,K] @ W + b) ----------
// One wave owns (1 mt x NTW nt); blockIdx.y selects the N-window for wide Nout.
// Inner loop is branch-free (last k-tile peeled, A prefetch unconditional); B
// fragments load in groups of 4 (one 8x b128 clause) so the scheduler can use
// graduated s_wait_loadcnt while earlier wmmas execute.
template <int NTW>
struct GemmBody {
  union Frag { v16bf v; uint4 q[2]; };

  static __device__ __forceinline__ void loadA(Frag& A, const unsigned short* abase,
                                               int kt) {
    const uint4* ap = (const uint4*)(abase + kt * 32);
    A.q[0] = ap[0];
    A.q[1] = ap[2];
  }

  static __device__ __forceinline__ void burst(const Frag& A, const uint4* bp,
                                               v8f* acc) {
#pragma unroll
    for (int jg = 0; jg < NTW; jg += 4) {
      Frag B0, B1, B2, B3;
      B0.q[0] = bp[(jg + 0) * 64 + 0]; B0.q[1] = bp[(jg + 0) * 64 + 1];
      B1.q[0] = bp[(jg + 1) * 64 + 0]; B1.q[1] = bp[(jg + 1) * 64 + 1];
      B2.q[0] = bp[(jg + 2) * 64 + 0]; B2.q[1] = bp[(jg + 2) * 64 + 1];
      B3.q[0] = bp[(jg + 3) * 64 + 0]; B3.q[1] = bp[(jg + 3) * 64 + 1];
#if defined(__gfx1250__)
      acc[jg + 0] = __builtin_amdgcn_wmma_f32_16x16x32_bf16(
          false, A.v, false, B0.v, (short)0, acc[jg + 0], false, false);
      acc[jg + 1] = __builtin_amdgcn_wmma_f32_16x16x32_bf16(
          false, A.v, false, B1.v, (short)0, acc[jg + 1], false, false);
      acc[jg + 2] = __builtin_amdgcn_wmma_f32_16x16x32_bf16(
          false, A.v, false, B2.v, (short)0, acc[jg + 2], false, false);
      acc[jg + 3] = __builtin_amdgcn_wmma_f32_16x16x32_bf16(
          false, A.v, false, B3.v, (short)0, acc[jg + 3], false, false);
#else
      (void)B0; (void)B1; (void)B2; (void)B3;
#endif
    }
  }
};

template <int NTW>
__global__ void __launch_bounds__(128)
k_gemm_row(const unsigned short* __restrict__ feat, const unsigned* __restrict__ Wp,
           const float* __restrict__ bias, int K, int Nout, float* __restrict__ out) {
  using Body = GemmBody<NTW>;
  using Frag = typename Body::Frag;

  int wave = threadIdx.x >> 5;
  int lane = threadIdx.x & 31;
  int mt = blockIdx.x * 4 + wave;
  if (mt >= N_NODES / 16) return;
  int nT = Nout >> 4;               // total n-tiles
  int ntBase = blockIdx.y * NTW;    // this wave's n-tile window
  int m0 = mt * 16;
  int mrow = m0 + (lane & 15);
  int hi = lane >> 4;

  v8f acc[NTW];
#pragma unroll
  for (int j = 0; j < NTW; ++j) acc[j] = (v8f){0.f, 0.f, 0.f, 0.f, 0.f, 0.f, 0.f, 0.f};

  const unsigned short* abase = feat + (long)mrow * K + hi * 8;
  const uint4* bcol = (const uint4*)Wp + ((long)ntBase * 64 + (long)lane * 2);
  long bktStride = (long)nT * 64;   // uint4 stride per k-tile
  int kTiles = K >> 5;

  Frag A;
  Body::loadA(A, abase, 0);

  // main loop: branch-free, A prefetched one k-tile ahead
  for (int kt = 0; kt < kTiles - 1; ++kt) {
    Frag Anext;
    Body::loadA(Anext, abase, kt + 1);
    Body::burst(A, bcol + kt * bktStride, acc);
    A = Anext;
  }
  // peeled last k-tile
  Body::burst(A, bcol + (long)(kTiles - 1) * bktStride, acc);

  int nlo = lane & 15;
#pragma unroll
  for (int j = 0; j < NTW; ++j) {
    int n = (ntBase + j) * 16 + nlo;
    float bn = bias[n];
#pragma unroll
    for (int r = 0; r < 8; ++r) {
      int m = m0 + hi * 8 + r;      // C/D layout: vgpr r, lanes 16-31 hold M+8
      out[(long)m * Nout + n] = fmaxf(acc[j][r] + bn, 0.0f);
    }
  }
}

// ---------- batchnorm ----------
__global__ void k_bn_stats(const float* __restrict__ h, int Nout,
                           float* __restrict__ mean, float* __restrict__ var) {
  __shared__ float ss[256], sq[256];
  int j = blockIdx.x;
  float s = 0.f, q = 0.f;
  for (int i = threadIdx.x; i < N_NODES; i += blockDim.x) {
    float v = h[(long)i * Nout + j];
    s += v; q += v * v;
  }
  ss[threadIdx.x] = s; sq[threadIdx.x] = q;
  __syncthreads();
  for (int o = blockDim.x >> 1; o > 0; o >>= 1) {
    if (threadIdx.x < o) { ss[threadIdx.x] += ss[threadIdx.x + o]; sq[threadIdx.x] += sq[threadIdx.x + o]; }
    __syncthreads();
  }
  if (threadIdx.x == 0) {
    float m = ss[0] * (1.0f / (float)N_NODES);
    mean[j] = m;
    var[j]  = sq[0] * (1.0f / (float)N_NODES) - m * m;
  }
}
__global__ void k_bn_apply(const float* __restrict__ h,
                           const float* __restrict__ mean, const float* __restrict__ var,
                           const float* __restrict__ gamma, const float* __restrict__ beta,
                           int sh, int Nout, float* __restrict__ out) {
  long idx = (long)blockIdx.x * blockDim.x + threadIdx.x;
  long total = (long)N_NODES << sh;
  if (idx >= total) return;
  int j = (int)(idx & (Nout - 1));
  float inv = rsqrtf(var[j] + EPS_F);
  out[idx] = (h[idx] - mean[j]) * inv * gamma[j] + beta[j];
}

// ---------- classifier: [N,64] @ [64,10] + bc ----------
__global__ void k_cls(const float* __restrict__ h, const float* __restrict__ Wc,
                      const float* __restrict__ bc, float* __restrict__ out) {
  int i = blockIdx.x * blockDim.x + threadIdx.x;
  if (i >= N_NODES) return;
  float acc[NCLS];
#pragma unroll
  for (int c = 0; c < NCLS; ++c) acc[c] = bc[c];
  const float* row = h + (long)i * 64;
  for (int k = 0; k < 64; ++k) {
    float v = row[k];
#pragma unroll
    for (int c = 0; c < NCLS; ++c) acc[c] += v * Wc[k * NCLS + c];
  }
#pragma unroll
  for (int c = 0; c < NCLS; ++c) out[(long)i * NCLS + c] = acc[c];
}

// ---------- host orchestration ----------
extern "C" void kernel_launch(void* const* d_in, const int* in_sizes, int n_in,
                              void* d_out, int out_size, void* d_ws, size_t ws_size,
                              hipStream_t stream) {
  (void)in_sizes; (void)n_in; (void)out_size; (void)ws_size;
  const float* x   = (const float*)d_in[0];
  const int*   ei  = (const int*)d_in[1];
  const int* srcp  = ei;
  const int* dstp  = ei + N_EDGES;
  const float* Wl[4] = {(const float*)d_in[2],  (const float*)d_in[6],
                        (const float*)d_in[10], (const float*)d_in[14]};
  const float* bl[4] = {(const float*)d_in[3],  (const float*)d_in[7],
                        (const float*)d_in[11], (const float*)d_in[15]};
  const float* gl[4] = {(const float*)d_in[4],  (const float*)d_in[8],
                        (const float*)d_in[12], (const float*)d_in[16]};
  const float* el[4] = {(const float*)d_in[5],  (const float*)d_in[9],
                        (const float*)d_in[13], (const float*)d_in[17]};
  const float* Wc = (const float*)d_in[18];
  const float* bc = (const float*)d_in[19];

  // workspace carve-out
  char* ws = (char*)d_ws;
  size_t off = 0;
  auto carve = [&](size_t bytes) -> char* {
    char* p = ws + off;
    off = (off + bytes + 255) & ~(size_t)255;
    return p;
  };
  const long MAXD = 256;
  float* deg    = (float*)carve((size_t)N_NODES * 4);
  float* logdeg = (float*)carve((size_t)N_NODES * 4);
  float* dsum   = (float*)carve(256);
  float* sum    = (float*)carve((size_t)N_NODES * MAXD * 4);
  float* sumsq  = (float*)carve((size_t)N_NODES * MAXD * 4);
  float* mnb    = (float*)carve((size_t)N_NODES * MAXD * 4);
  float* mxb    = (float*)carve((size_t)N_NODES * MAXD * 4);
  unsigned short* feat = (unsigned short*)carve((size_t)N_NODES * 13 * MAXD * 2);
  float* hpre   = (float*)carve((size_t)N_NODES * MAXD * 4);
  float* hA     = (float*)carve((size_t)N_NODES * MAXD * 4);
  float* hB     = (float*)carve((size_t)N_NODES * MAXD * 4);
  unsigned* Wp  = (unsigned*)carve((size_t)13 * MAXD * MAXD * 2);
  float* bnm    = (float*)carve((size_t)MAXD * 4);
  float* bnv    = (float*)carve((size_t)MAXD * 4);

  // degree, log1p(deg), delta numerator
  k_zero_f<<<(N_NODES + 255) / 256, 256, 0, stream>>>(deg, N_NODES);
  k_zero_f<<<1, 1, 0, stream>>>(dsum, 1);
  k_deg<<<(N_EDGES + 255) / 256, 256, 0, stream>>>(dstp, deg);
  k_logdeg<<<(N_NODES + 255) / 256, 256, 0, stream>>>(deg, logdeg, dsum);

  const int din_[4]  = {64, 128, 256, 128};
  const int dout_[4] = {128, 256, 128, 64};
  const float* cur = x;
  float* nxt[4] = {hA, hB, hA, hB};

  const int mTiles = N_NODES / 16;            // 3125 (exact)
  const int gemmBlocks = (mTiles + 3) / 4;    // 4 waves / block

  for (int L = 0; L < 4; ++L) {
    int d = din_[L], dn = dout_[L], ldf = 13 * d;
    int sh = __builtin_ctz(d), shn = __builtin_ctz(dn);
    long nd = (long)N_NODES * d;
    long ed = (long)N_EDGES * d;

    k_zero_f<<<(int)((nd + 255) / 256), 256, 0, stream>>>(sum, nd);
    k_zero_f<<<(int)((nd + 255) / 256), 256, 0, stream>>>(sumsq, nd);
    k_fill_u<<<(int)((nd + 255) / 256), 256, 0, stream>>>((unsigned*)mnb, nd, 0x7F800000u);
    k_fill_u<<<(int)((nd + 255) / 256), 256, 0, stream>>>((unsigned*)mxb, nd, 0xFF800000u);

    k_scatter<<<(int)((ed + 255) / 256), 256, 0, stream>>>(
        cur, srcp, dstp, sh, d, sum, sumsq, mnb, mxb);

    k_build_feat<<<(int)((nd + 255) / 256), 256, 0, stream>>>(
        cur, deg, logdeg, dsum, sh, d, ldf, sum, sumsq, mnb, mxb, feat);

    long wn = (long)ldf * dn / 2;
    k_packW<<<(int)((wn + 255) / 256), 256, 0, stream>>>(Wl[L], ldf, dn, Wp);

    if (dn == 256) {
      dim3 g(gemmBlocks, 2);   // two N-halves of 8 tiles each
      k_gemm_row<8><<<g, 128, 0, stream>>>(feat, Wp, bl[L], ldf, dn, hpre);
    } else if (dn == 128) {
      dim3 g(gemmBlocks, 1);
      k_gemm_row<8><<<g, 128, 0, stream>>>(feat, Wp, bl[L], ldf, dn, hpre);
    } else {
      dim3 g(gemmBlocks, 1);
      k_gemm_row<4><<<g, 128, 0, stream>>>(feat, Wp, bl[L], ldf, dn, hpre);
    }

    k_bn_stats<<<dn, 256, 0, stream>>>(hpre, dn, bnm, bnv);
    long no = (long)N_NODES * dn;
    k_bn_apply<<<(int)((no + 255) / 256), 256, 0, stream>>>(
        hpre, bnm, bnv, gl[L], el[L], shn, dn, nxt[L]);

    cur = nxt[L];
  }

  k_cls<<<(N_NODES + 255) / 256, 256, 0, stream>>>(cur, Wc, bc, (float*)d_out);
}
